// MixtureRNN_40656160424375
// MI455X (gfx1250) — compile-verified, matrix operand order checked by reading, not run on hardware
//
#include <hip/hip_runtime.h>
#include <cstdint>
#include <cstddef>

#define B_ 256
#define T_ 256
#define E_ 512
#define M_ 16
#define V_ 32000

typedef __bf16 bf16;
typedef bf16  bf16x16 __attribute__((ext_vector_type(16)));
typedef float floatx8 __attribute__((ext_vector_type(8)));

union AFrag { uint4 u[2]; bf16x16 v; bf16 h[16]; };
union CFrag { floatx8 v; float f[8]; };

// k_local (0..31) for WMMA bf16 A/B fragment element e of a given lane
// (CDNA5 ISA 7.12.2: 16-bit A 16x32 layout; B symmetric with N<->M).
__device__ __forceinline__ int frag_k(int lane, int e) {
    int r = e >> 1, h = e & 1;
    return ((r >> 2) << 4) + ((lane & 16) ? 8 : 0) + ((r & 3) << 1) + h;
}

// ---------------------------------------------------------------------------
// Prep 1: convert Wq/Wk/Wv (f32 row-major KxN) to bf16 in WMMA-fragment order:
// out[mat][nt*16+kt][lane][e], so the GEMM loads 2 contiguous b128 per frag.
// ---------------------------------------------------------------------------
__global__ void swizzle_weights(const float* __restrict__ Wq,
                                const float* __restrict__ Wk,
                                const float* __restrict__ Wv,
                                bf16* __restrict__ out) {
    int gid = blockIdx.x * blockDim.x + threadIdx.x;   // 3*512*512 = 786432
    int mat = gid >> 18;
    int rem = gid & 262143;
    int nt  = rem >> 13;
    int r2  = rem & 8191;
    int kt  = r2 >> 9;
    int r3  = r2 & 511;
    int lane = r3 >> 4;
    int e    = r3 & 15;
    int n = nt * 16 + (lane & 15);
    int k = kt * 32 + frag_k(lane, e);
    const float* W = (mat == 0) ? Wq : (mat == 1) ? Wk : Wv;
    out[gid] = (bf16)W[k * E_ + n];
}

// ---------------------------------------------------------------------------
// Prep 2: Km = mover_w @ Wk + bk, Vm = mover_w @ Wv + bv  (16x512 each, f32).
// One-time 8.4 MFLOP — plain VALU is fine.
// ---------------------------------------------------------------------------
__global__ void mover_kv(const float* __restrict__ mover,
                         const float* __restrict__ Wk, const float* __restrict__ bk,
                         const float* __restrict__ Wv, const float* __restrict__ bv,
                         float* __restrict__ Km, float* __restrict__ Vm) {
    int gid = blockIdx.x * blockDim.x + threadIdx.x;   // 8192
    int m = gid >> 9, n = gid & 511;
    float aK = bk[n], aV = bv[n];
    for (int k = 0; k < E_; ++k) {
        float mv = mover[m * E_ + k];
        aK += mv * Wk[k * E_ + n];
        aV += mv * Wv[k * E_ + n];
    }
    Km[gid] = aK;
    Vm[gid] = aV;
}

// ---------------------------------------------------------------------------
// Scan: 16 persistent workgroups, each owns a 16-row batch tile (WMMA M=16).
// Per step: z += sqrt(E)*emb[idx]; q/k/v tile GEMMs (bf16 WMMA, f32 accum);
// 17-way softmax attention vs [kz; Km]; residual + LayerNorm.
// Block = 512 threads = 16 waves (4/SIMD -> ~128 VGPR budget, no spills).
// Each wave computes 6 of the 96 (mat, n-tile) output tiles sequentially.
// K-tile order is rotated by (t + tile) so weight-fragment loads cannot be
// hoisted out of the step loop (which previously forced scratch spills).
// ---------------------------------------------------------------------------
__launch_bounds__(512)
__global__ void scan_step(const float* __restrict__ hidden,
                          const int*   __restrict__ seq,
                          const float* __restrict__ emb,
                          const bf16*  __restrict__ wF,
                          const float* __restrict__ Km,
                          const float* __restrict__ Vm,
                          const float* __restrict__ bq,
                          const float* __restrict__ bk,
                          const float* __restrict__ bv,
                          const float* __restrict__ gamma,
                          const float* __restrict__ beta,
                          float* __restrict__ zOut,
                          bf16*  __restrict__ zfrag) {
    __shared__ __align__(16) unsigned char smem[65536];
    bf16 (*zbf)[E_] = reinterpret_cast<bf16(*)[E_]>(smem);           // 16KB A-staging
    bf16 (*qs)[E_]  = reinterpret_cast<bf16(*)[E_]>(smem + 16384);   // 16KB
    bf16 (*ks)[E_]  = reinterpret_cast<bf16(*)[E_]>(smem + 32768);   // 16KB
    bf16 (*vs)[E_]  = reinterpret_cast<bf16(*)[E_]>(smem + 49152);   // 16KB
    // after scores are consumed, qs+ks (32KB) are reused as f32 z_new (16x512)
    float* znew = reinterpret_cast<float*>(smem + 16384);

    const int tid  = threadIdx.x;
    const int wave = tid >> 5;
    const int lane = tid & 31;
    const int bt   = blockIdx.x;
    const bool worker = tid < 256;
    const int r = tid >> 4;            // row in tile  (workers only)
    const int c = tid & 15;            // column class (workers only)
    const int bg = bt * 16 + r;

    const float sqrtE = 22.62741699796952f;
    const float scale = 0.04419417382415922f;   // 1/sqrt(512)

    float zr[32];                      // worker-owned z slice: e = c + 16*i
    if (worker) {
#pragma unroll
        for (int i = 0; i < 32; ++i)
            zr[i] = hidden[(size_t)bg * E_ + c + 16 * i];
    }
    float att[17];

    for (int t = 0; t < T_; ++t) {
        // ---- Phase A: z += sqrt(E) * emb_out[seq[b,t]]; refresh bf16 copy
        if (worker) {
            int idx = seq[bg * T_ + t];
            const float* er = emb + (size_t)idx * E_;
#pragma unroll
            for (int i = 0; i < 32; ++i) {
                int e = c + 16 * i;
                float z = zr[i] + er[e] * sqrtE;
                zr[i] = z;
                zbf[r][e] = (bf16)z;
            }
        }
        __syncthreads();

        // ---- Phase B: q/k/v = z @ W (+bias); 96 tiles over 16 waves
        {
            const int koff = (lane & 16) ? 8 : 0;
            const int am   = lane & 15;
            for (int tile = wave; tile < 96; tile += 16) {
                const int mat = tile >> 5;        // 0=q, 1=k, 2=v
                const int nt  = tile & 31;
                const bf16* wBase = wF + ((size_t)mat << 18);

                CFrag acc;
#pragma unroll
                for (int v = 0; v < 8; ++v) acc.f[v] = 0.f;

#pragma unroll 4
                for (int kt = 0; kt < 16; ++kt) {
                    const int ktr = (kt + t + tile) & 15;  // rotation: no LICM
                    AFrag a;
                    const uint4* pa = reinterpret_cast<const uint4*>(
                        &zbf[am][ktr * 32 + koff]);
                    a.u[0] = pa[0];
                    a.u[1] = pa[2];     // second 8-element K chunk (+16 elems)
                    AFrag b;
                    const uint4* pb = reinterpret_cast<const uint4*>(
                        wBase + ((size_t)(nt * 16 + ktr) << 9) +
                        ((size_t)lane << 4));
                    b.u[0] = pb[0];
                    b.u[1] = pb[1];
                    acc.v = __builtin_amdgcn_wmma_f32_16x16x32_bf16(
                        false, a.v, false, b.v, (short)0, acc.v, false, false);
                }

                bf16 (*dst)[E_]  = (mat == 0) ? qs : (mat == 1) ? ks : vs;
                const float* bp  = (mat == 0) ? bq : (mat == 1) ? bk : bv;
                const int n      = nt * 16 + (lane & 15);
                const int mbase  = (lane & 16) ? 8 : 0;
                const float bias = bp[n];
#pragma unroll
                for (int v = 0; v < 8; ++v)
                    dst[mbase + v][n] = (bf16)(acc.f[v] + bias);
            }
        }
        __syncthreads();

        // ---- Phase C1: 17 scores per row (q.kz, q.Km[0..15]) + softmax
        if (worker) {
            float s[17];
#pragma unroll
            for (int j = 0; j < 17; ++j) s[j] = 0.f;
            for (int i = 0; i < 32; ++i) {
                int e = c + 16 * i;
                float qv = (float)qs[r][e];
                s[0] += qv * (float)ks[r][e];
#pragma unroll
                for (int m = 0; m < 16; ++m) s[1 + m] += qv * Km[m * E_ + e];
            }
#pragma unroll
            for (int j = 0; j < 17; ++j) {
                float v = s[j];
                v += __shfl_xor(v, 1, 16);
                v += __shfl_xor(v, 2, 16);
                v += __shfl_xor(v, 4, 16);
                v += __shfl_xor(v, 8, 16);
                s[j] = v * scale;
            }
            float mx = s[0];
#pragma unroll
            for (int j = 1; j < 17; ++j) mx = fmaxf(mx, s[j]);
            float sum = 0.f;
#pragma unroll
            for (int j = 0; j < 17; ++j) {
                att[j] = expf(s[j] - mx);
                sum += att[j];
            }
            float inv = 1.f / sum;
#pragma unroll
            for (int j = 0; j < 17; ++j) att[j] *= inv;
        }
        __syncthreads();   // scores done -> safe to overlay znew on qs/ks

        // ---- Phase C2: dg0 = att.[vz; Vm]; residual; LayerNorm; refresh zbf
        if (worker) {
            float sum = 0.f, sq = 0.f;
            for (int i = 0; i < 32; ++i) {
                int e = c + 16 * i;
                float dv = att[0] * (float)vs[r][e];
#pragma unroll
                for (int m = 0; m < 16; ++m) dv += att[1 + m] * Vm[m * E_ + e];
                float z = zr[i] + dv;
                znew[r * E_ + e] = z;
                sum += z;
                sq  += z * z;
            }
            sum += __shfl_xor(sum, 1, 16);
            sum += __shfl_xor(sum, 2, 16);
            sum += __shfl_xor(sum, 4, 16);
            sum += __shfl_xor(sum, 8, 16);
            sq  += __shfl_xor(sq, 1, 16);
            sq  += __shfl_xor(sq, 2, 16);
            sq  += __shfl_xor(sq, 4, 16);
            sq  += __shfl_xor(sq, 8, 16);
            float mean = sum * (1.f / 512.f);
            float var  = sq * (1.f / 512.f) - mean * mean;
            float rstd = rsqrtf(var + 1e-6f);
            for (int i = 0; i < 32; ++i) {
                int e = c + 16 * i;
                float z = (znew[r * E_ + e] - mean) * rstd * gamma[e] + beta[e];
                zr[i] = z;
                zbf[r][e] = (bf16)z;
            }
        }
        __syncthreads();
    }

    // ---- Epilogue: final z (f32) to d_out; bf16 A-fragments of z to ws
    if (worker) {
#pragma unroll
        for (int i = 0; i < 32; ++i)
            zOut[(size_t)bg * E_ + c + 16 * i] = zr[i];
    }
    __syncthreads();
    for (int idx = tid; idx < 8192; idx += 512) {
        int kt  = idx >> 9;
        int rem = idx & 511;
        int ln  = rem >> 4;
        int e   = rem & 15;
        int k   = kt * 32 + frag_k(ln, e);
        zfrag[((size_t)(bt * 16 + kt) << 9) + rem] = zbf[ln & 15][k];
    }
}

// ---------------------------------------------------------------------------
// Vocab logits: (B x E) bf16 fragments @ (E x V) f32 (converted in-register)
// with bf16 WMMA; vocab_W is single-use so on-the-fly f32->bf16 is cheapest.
// grid (250, 16): 8 waves/block, each wave owns one 16-wide vocab tile.
// ---------------------------------------------------------------------------
__launch_bounds__(256)
__global__ void vocab_logits(const bf16*  __restrict__ zfrag,
                             const float* __restrict__ vocabW,
                             const float* __restrict__ vocabB,
                             float* __restrict__ yout) {
    const int tid  = threadIdx.x;
    const int wave = tid >> 5;
    const int lane = tid & 31;
    const int bt   = blockIdx.y;
    const int vt   = blockIdx.x * 8 + wave;          // 0..1999
    const int n    = vt * 16 + (lane & 15);

    CFrag acc;
#pragma unroll
    for (int v = 0; v < 8; ++v) acc.f[v] = 0.f;

    for (int kt = 0; kt < 16; ++kt) {
        AFrag a;
        const uint4* pa = reinterpret_cast<const uint4*>(
            zfrag + ((size_t)(bt * 16 + kt) << 9) + ((size_t)lane << 4));
        a.u[0] = pa[0];
        a.u[1] = pa[1];
        AFrag b;
#pragma unroll
        for (int e = 0; e < 16; ++e) {
            int k = kt * 32 + frag_k(lane, e);
            b.h[e] = (bf16)vocabW[(size_t)k * V_ + n];
        }
        acc.v = __builtin_amdgcn_wmma_f32_16x16x32_bf16(
            false, a.v, false, b.v, (short)0, acc.v, false, false);
    }

    const float bias  = vocabB[n];
    const int   mbase = (lane & 16) ? 8 : 0;
#pragma unroll
    for (int v = 0; v < 8; ++v) {
        int b_ = bt * 16 + mbase + v;
        yout[(size_t)b_ * V_ + n] = acc.f[v] + bias;
    }
}

// ---------------------------------------------------------------------------
// In-place log-softmax over each row of 32000 logits. One block per row.
// ---------------------------------------------------------------------------
__launch_bounds__(256)
__global__ void logsoftmax_rows(float* __restrict__ yout) {
    const int row = blockIdx.x;
    float* x = yout + (size_t)row * V_;
    const int tid = threadIdx.x;
    __shared__ float red[8];

    float m = -3.4e38f;
    for (int i = tid; i < V_; i += 256) m = fmaxf(m, x[i]);
#pragma unroll
    for (int o = 16; o > 0; o >>= 1) m = fmaxf(m, __shfl_xor(m, o, 32));
    if ((tid & 31) == 0) red[tid >> 5] = m;
    __syncthreads();
    if (tid == 0) {
        float mm = red[0];
        for (int i = 1; i < 8; ++i) mm = fmaxf(mm, red[i]);
        red[0] = mm;
    }
    __syncthreads();
    const float rowmax = red[0];
    __syncthreads();

    float s = 0.f;
    for (int i = tid; i < V_; i += 256) s += expf(x[i] - rowmax);
#pragma unroll
    for (int o = 16; o > 0; o >>= 1) s += __shfl_xor(s, o, 32);
    if ((tid & 31) == 0) red[tid >> 5] = s;
    __syncthreads();
    if (tid == 0) {
        float ss = 0.f;
        for (int i = 0; i < 8; ++i) ss += red[i];
        red[0] = logf(ss);
    }
    __syncthreads();
    const float lse = red[0];
    for (int i = tid; i < V_; i += 256) x[i] = x[i] - rowmax - lse;
}

// ---------------------------------------------------------------------------
extern "C" void kernel_launch(void* const* d_in, const int* in_sizes, int n_in,
                              void* d_out, int out_size, void* d_ws, size_t ws_size,
                              hipStream_t stream) {
    const float* hidden = (const float*)d_in[0];
    const int*   seq    = (const int*)  d_in[1];
    const float* emb    = (const float*)d_in[2];
    const float* Wq     = (const float*)d_in[3];
    const float* bq     = (const float*)d_in[4];
    const float* Wk     = (const float*)d_in[5];
    const float* bk     = (const float*)d_in[6];
    const float* Wv     = (const float*)d_in[7];
    const float* bv     = (const float*)d_in[8];
    const float* mover  = (const float*)d_in[9];
    const float* gamma  = (const float*)d_in[10];
    const float* beta   = (const float*)d_in[11];
    const float* vocW   = (const float*)d_in[12];
    const float* vocB   = (const float*)d_in[13];

    unsigned char* ws = (unsigned char*)d_ws;
    bf16*  wF    = (bf16*)(ws);               // 3 * 262144 bf16 = 1.5 MB
    float* Km    = (float*)(ws + 1572864);    // 16x512 f32
    float* Vm    = (float*)(ws + 1605632);    // 16x512 f32
    bf16*  zfrag = (bf16*)(ws + 1638400);     // 16x16x512 bf16 fragments

    float* zOut = (float*)d_out;              // (B,1,E)
    float* yOut = zOut + (size_t)B_ * E_;     // (B,1,V)

    swizzle_weights<<<3072, 256, 0, stream>>>(Wq, Wk, Wv, wF);
    mover_kv<<<32, 256, 0, stream>>>(mover, Wk, bk, Wv, bv, Km, Vm);
    scan_step<<<16, 512, 0, stream>>>(hidden, seq, emb, wF, Km, Vm,
                                      bq, bk, bv, gamma, beta, zOut, zfrag);
    vocab_logits<<<dim3(250, 16), 256, 0, stream>>>(zfrag, vocW, vocB, yOut);
    logsoftmax_rows<<<256, 256, 0, stream>>>(yOut);
}
